// Joint_14164802142386
// MI455X (gfx1250) — matrix operand order repaired and checked by reading, not compiled, not run on hardware
//
#include <hip/hip_runtime.h>
#include <hip/hip_bf16.h>

// Problem constants (fixed by the reference).
#define B_      8
#define T_      200
#define UU_     101
#define ENC_H   1024
#define PRED_H  320
#define JOINT_H 512
#define VOCAB   1024
#define IN_DIM  (ENC_H + PRED_H)

// ---- CDNA5 WMMA types ------------------------------------------------------
typedef __attribute__((ext_vector_type(16))) __bf16 v16bf;
typedef __attribute__((ext_vector_type(8)))  float  v8f;

struct alignas(16) U4 { unsigned x, y, z, w; };
union FragBF { v16bf v; U4 q[2]; };   // 32 bytes = 8 VGPRs = one A/B fragment

__device__ __forceinline__ unsigned short f32_to_bf16(float f) {
  union { float f; unsigned u; } x; x.f = f;
  unsigned r = x.u + 0x7FFFu + ((x.u >> 16) & 1u);  // round-to-nearest-even
  return (unsigned short)(r >> 16);
}

// ---- Kernel 1: W2 (fp32, [VOCAB][JOINT_H]) -> bf16 workspace ---------------
__global__ __launch_bounds__(256) void cvt_w2_bf16(const float* __restrict__ W2,
                                                   unsigned short* __restrict__ W2b) {
  const int i = (blockIdx.x * 256 + threadIdx.x) * 4;
  if (i < VOCAB * JOINT_H) {
    float4 v = *(const float4*)(W2 + i);
    uint2 pk;
    pk.x = (unsigned)f32_to_bf16(v.x) | ((unsigned)f32_to_bf16(v.y) << 16);
    pk.y = (unsigned)f32_to_bf16(v.z) | ((unsigned)f32_to_bf16(v.w) << 16);
    *(uint2*)(W2b + i) = pk;
  }
}

// ---- Kernel 2: pf[b,t,:] = f[b,t,:] @ W1[:, :ENC_H]^T (fp32) ---------------
__global__ __launch_bounds__(256) void proj_enc(const float* __restrict__ f,
                                                const float* __restrict__ W1,
                                                float* __restrict__ pf) {
  __shared__ float sf[ENC_H];
  const int bt = blockIdx.x;                       // 0 .. B*T-1
  const float* fr = f + (size_t)bt * ENC_H;
  for (int i = threadIdx.x; i < ENC_H; i += 256) sf[i] = fr[i];
  __syncthreads();
  for (int j = threadIdx.x; j < JOINT_H; j += 256) {
    const float* w = W1 + (size_t)j * IN_DIM;
    float s = 0.f;
#pragma unroll 4
    for (int k = 0; k < ENC_H; k += 4) {
      float4 wv = *(const float4*)(w + k);
      s = fmaf(wv.x, sf[k + 0], s);
      s = fmaf(wv.y, sf[k + 1], s);
      s = fmaf(wv.z, sf[k + 2], s);
      s = fmaf(wv.w, sf[k + 3], s);
    }
    pf[(size_t)bt * JOINT_H + j] = s;
  }
}

// ---- Kernel 3: pg[b,u,:] = g[b,u,:] @ W1[:, ENC_H:]^T (fp32) ---------------
__global__ __launch_bounds__(256) void proj_pred(const float* __restrict__ g,
                                                 const float* __restrict__ W1,
                                                 float* __restrict__ pg) {
  __shared__ float sg[PRED_H];
  const int bu = blockIdx.x;                       // 0 .. B*U-1
  const float* gr = g + (size_t)bu * PRED_H;
  for (int i = threadIdx.x; i < PRED_H; i += 256) sg[i] = gr[i];
  __syncthreads();
  for (int j = threadIdx.x; j < JOINT_H; j += 256) {
    const float* w = W1 + (size_t)j * IN_DIM + ENC_H;
    float s = 0.f;
#pragma unroll 4
    for (int k = 0; k < PRED_H; k += 4) {
      float4 wv = *(const float4*)(w + k);
      s = fmaf(wv.x, sg[k + 0], s);
      s = fmaf(wv.y, sg[k + 1], s);
      s = fmaf(wv.z, sg[k + 2], s);
      s = fmaf(wv.w, sg[k + 3], s);
    }
    pg[(size_t)bu * JOINT_H + j] = s;
  }
}

// ---- Kernel 4: fused h = relu(pf+pg+b1) -> bf16 LDS tile -> WMMA GEMM ------
// Per workgroup: 64 rows (flattened t*U+u within one batch) x 512 vocab cols.
// 8 waves; wave w owns columns [nh*512 + w*64, +64): 4 M-subtiles x 4 N-subtiles.
#define MROWS   64
#define LDK     (JOINT_H + 8)                       // ushort row pitch (pad 16B)
#define MT_PER_B ((T_ * UU_ + MROWS - 1) / MROWS)   // 316

__global__ __launch_bounds__(256) void rnnt_joint_wmma(
    const float* __restrict__ pf, const float* __restrict__ pg,
    const float* __restrict__ b1, const unsigned short* __restrict__ W2b,
    const float* __restrict__ b2, float* __restrict__ out) {

  __shared__ __align__(16) unsigned short shA[MROWS * LDK];  // 66,560 B bf16 bits

  const int blk = blockIdx.x;
  const int nh  = blk & 1;                  // which 512-wide half of VOCAB
  const int mt  = (blk >> 1) % MT_PER_B;    // 64-row tile within batch
  const int b   = blk / (2 * MT_PER_B);
  const int tid = threadIdx.x;

  // ---- Stage h tile: h[r][k] = relu(pf[b,t][k] + pg[b,u][k] + b1[k]) as bf16
  for (int i = tid; i < MROWS * (JOINT_H / 4); i += 256) {
    const int row = i >> 7;                 // JOINT_H/4 == 128
    const int k4  = (i & 127) << 2;
    const int rg  = mt * MROWS + row;       // flattened t*U + u
    uint2 pk; pk.x = 0u; pk.y = 0u;
    if (rg < T_ * UU_) {
      const int t = rg / UU_;
      const int u = rg % UU_;
      float4 a = *(const float4*)(pf + ((size_t)b * T_  + t) * JOINT_H + k4);
      float4 q = *(const float4*)(pg + ((size_t)b * UU_ + u) * JOINT_H + k4);
      float4 c = *(const float4*)(b1 + k4);
      float h0 = fmaxf(a.x + q.x + c.x, 0.f);
      float h1 = fmaxf(a.y + q.y + c.y, 0.f);
      float h2 = fmaxf(a.z + q.z + c.z, 0.f);
      float h3 = fmaxf(a.w + q.w + c.w, 0.f);
      pk.x = (unsigned)f32_to_bf16(h0) | ((unsigned)f32_to_bf16(h1) << 16);
      pk.y = (unsigned)f32_to_bf16(h2) | ((unsigned)f32_to_bf16(h3) << 16);
    }
    *(uint2*)&shA[row * LDK + k4] = pk;
  }
  __syncthreads();

  // ---- WMMA: 4 M-subtiles x 4 N-subtiles per wave over K=512 (16 steps) ----
  const int wave = tid >> 5;
  const int lane = tid & 31;
  const int half = lane >> 4;               // K-half select per ISA A/B layout
  const int l15  = lane & 15;
  const int n0   = nh * 512 + wave * 64;

  // A: lane row = mi*16+l15; VGPR0-3 = K(+half*8)+0..7, VGPR4-7 = +16
  const unsigned short* ap = &shA[l15 * LDK + half * 8];
  // B: lane column = n0+ni*16+l15; VGPR0-3 = K(+half*8)+0..7, VGPR4-7 = +16
  const unsigned short* bp = W2b + ((size_t)(n0 + l15) * JOINT_H) + half * 8;

  v8f acc[4][4] = {};
  for (int ks = 0; ks < JOINT_H; ks += 32) {
    FragBF A[4];
#pragma unroll
    for (int mi = 0; mi < 4; ++mi) {
      const unsigned short* p = ap + mi * 16 * LDK + ks;
      A[mi].q[0] = *(const U4*)(p);         // ds_load_b128
      A[mi].q[1] = *(const U4*)(p + 16);
    }
#pragma unroll
    for (int ni = 0; ni < 4; ++ni) {
      FragBF Bf;
      const unsigned short* q = bp + (size_t)ni * 16 * JOINT_H + ks;
      Bf.q[0] = *(const U4*)(q);            // global_load_b128 (L2-resident W2)
      Bf.q[1] = *(const U4*)(q + 16);
#pragma unroll
      for (int mi = 0; mi < 4; ++mi) {
        acc[mi][ni] = __builtin_amdgcn_wmma_f32_16x16x32_bf16(
            false, A[mi].v, false, Bf.v, (short)0, acc[mi][ni], false, false);
      }
    }
  }

  // ---- Epilogue: + b2, masked store (out idx = (b*T*U + rg)*VOCAB + n) -----
  const size_t outBase = (size_t)b * T_ * UU_;
#pragma unroll
  for (int ni = 0; ni < 4; ++ni) {
    const int n = n0 + ni * 16 + l15;
    const float bias = b2[n];
#pragma unroll
    for (int mi = 0; mi < 4; ++mi) {
#pragma unroll
      for (int r = 0; r < 8; ++r) {
        const int m  = mi * 16 + r + half * 8;   // C/D layout: VGPR r -> M=r(+8)
        const int rg = mt * MROWS + m;
        if (rg < T_ * UU_) {
          out[(outBase + rg) * VOCAB + n] = acc[mi][ni][r] + bias;
        }
      }
    }
  }
}

// ---- Host-side launcher ----------------------------------------------------
extern "C" void kernel_launch(void* const* d_in, const int* in_sizes, int n_in,
                              void* d_out, int out_size, void* d_ws, size_t ws_size,
                              hipStream_t stream) {
  (void)in_sizes; (void)n_in; (void)out_size; (void)ws_size;
  const float* f  = (const float*)d_in[0];
  const float* g  = (const float*)d_in[1];
  const float* W1 = (const float*)d_in[2];
  const float* b1 = (const float*)d_in[3];
  const float* W2 = (const float*)d_in[4];
  const float* b2 = (const float*)d_in[5];
  float* out = (float*)d_out;

  // Workspace layout: pf (3.28 MB) | pg (1.65 MB) | W2 bf16 (1 MB)
  float* pf = (float*)d_ws;
  float* pg = pf + (size_t)B_ * T_ * JOINT_H;
  unsigned short* W2b = (unsigned short*)(pg + (size_t)B_ * UU_ * JOINT_H);

  cvt_w2_bf16<<<(VOCAB * JOINT_H) / (256 * 4), 256, 0, stream>>>(W2, W2b);
  proj_enc <<<B_ * T_,  256, 0, stream>>>(f, W1, pf);
  proj_pred<<<B_ * UU_, 256, 0, stream>>>(g, W1, pg);
  rnnt_joint_wmma<<<B_ * MT_PER_B * 2, 256, 0, stream>>>(pf, pg, b1, W2b, b2, out);
}
